// LapVAE_79731772883375
// MI455X (gfx1250) — compile-verified, matrix operand order checked by reading, not run on hardware
//
#include <hip/hip_runtime.h>
#include <hip/hip_bf16.h>

// ---------------------------------------------------------------------------
// LapVAE forward for MI455X (gfx1250).  All heavy linear layers run on the
// WMMA pipe (v_wmma_f32_16x16x32_bf16) with BN folded into the weights.
// ---------------------------------------------------------------------------

#define BB   64
#define NN   5023
#define FF   64
#define DLAT 8
#define NNZ  (NN * 8)
#define MROWS ((size_t)BB * NN)          // 321472, divisible by 32
#define EPS_BN 1e-5f

typedef __attribute__((ext_vector_type(16))) __bf16 v16bf;
typedef __attribute__((ext_vector_type(16))) float  v16f;
typedef __attribute__((ext_vector_type(8)))  float  v8f;
typedef __attribute__((ext_vector_type(4)))  float  v4f;

__device__ __forceinline__ __bf16 f2bf(float f) {
    return (__bf16)f;                     // native cvt, RNE in hardware
}

__device__ __forceinline__ float elu1(float v) {
    return v > 0.f ? v : (__expf(v) - 1.f);
}

// ---------------------------------------------------------------------------
// WMMA GEMM:  Y[M x 64] = X[M x K] * W' + b'  (+ residual) (+ elu)
// W' pre-packed in bf16 B-fragment layout by k_fold.
// Block: 256 thr = 8 waves; wave -> one 16x16 tile; block -> 32 rows x 64 cols.
// K, residual-add, and elu are compile-time -> fully unrolled, branch-free.
// Grid must be M/32 with M a multiple of 32 (true for all call sites).
// ---------------------------------------------------------------------------
template<int K, bool RES, bool ELU>
__global__ void k_gemm_wmma(const float* __restrict__ X,
                            const __bf16* __restrict__ Wfrag,
                            const float* __restrict__ bp,
                            const float* __restrict__ res,
                            float* __restrict__ Y)
{
    constexpr int kcs = K >> 5;               // k-chunks of 32
    const int lane    = threadIdx.x & 31;
    const int wave    = threadIdx.x >> 5;
    const int rowTile = wave >> 2;            // 0..1
    const int colTile = wave & 3;             // 0..3
    const int rowBase = blockIdx.x * 32 + rowTile * 16;
    const int mrow    = lane & 15;
    const int khalf   = (lane >> 4) * 8;      // lanes 16..31 start at K+8

    v8f acc = {};
    const float* xrow = X + (size_t)(rowBase + mrow) * K;

#pragma unroll
    for (int c = 0; c < kcs; ++c) {
        // ---- A fragment (16x32 bf16, per ISA lane layout) ----
        const float* p0 = xrow + c * 32 + khalf;        // K offsets +0..7
        const float* p1 = p0 + 16;                      // K offsets +16..23
        if (c + 1 < kcs) __builtin_prefetch(p0 + 32, 0, 1);   // global_prefetch_b8
        v4f a0 = *(const v4f*)(p0);
        v4f a1 = *(const v4f*)(p0 + 4);
        v4f b0 = *(const v4f*)(p1);
        v4f b1 = *(const v4f*)(p1 + 4);
        v16f xf;
        xf[0]  = a0[0]; xf[1]  = a0[1]; xf[2]  = a0[2]; xf[3]  = a0[3];
        xf[4]  = a1[0]; xf[5]  = a1[1]; xf[6]  = a1[2]; xf[7]  = a1[3];
        xf[8]  = b0[0]; xf[9]  = b0[1]; xf[10] = b0[2]; xf[11] = b0[3];
        xf[12] = b1[0]; xf[13] = b1[1]; xf[14] = b1[2]; xf[15] = b1[3];
        v16bf a = __builtin_convertvector(xf, v16bf);   // packed v_cvt -> bf16
        // ---- B fragment: pre-packed, one contiguous 32B load per lane ----
        v16bf b = *(const v16bf*)(Wfrag + ((size_t)((colTile * kcs + c) * 32 + lane)) * 16);
        acc = __builtin_amdgcn_wmma_f32_16x16x32_bf16(
                  false, a, false, b, (short)0, acc, false, false);
    }

    // ---- epilogue: bias + residual + elu, ISA C/D layout (branch-free) ----
    const int col  = colTile * 16 + (lane & 15);
    const int mofs = (lane < 16) ? 0 : 8;
    const float bias = bp[col];
#pragma unroll
    for (int r = 0; r < 8; ++r) {
        const size_t ofs = (size_t)(rowBase + mofs + r) * 64 + col;
        float v = acc[r] + bias;
        if constexpr (RES) v += res[ofs];
        if constexpr (ELU) v = elu1(v);
        Y[ofs] = v;
    }
}

static void launch_gemm(const float* X, const __bf16* Wf, const float* bp,
                        const float* res, float* Y, size_t M, int K, bool elu,
                        hipStream_t st)
{
    const unsigned g = (unsigned)(M / 32);
    if (K == 128) {
        if (res) k_gemm_wmma<128, true,  false><<<g, 256, 0, st>>>(X, Wf, bp, res, Y);
        else     k_gemm_wmma<128, false, false><<<g, 256, 0, st>>>(X, Wf, bp, nullptr, Y);
    } else {
        if (elu) k_gemm_wmma<64, false, true ><<<g, 256, 0, st>>>(X, Wf, bp, nullptr, Y);
        else     k_gemm_wmma<64, false, false><<<g, 256, 0, st>>>(X, Wf, bp, nullptr, Y);
    }
}

// ---------------------------------------------------------------------------
// Per-channel batch stats: sums[c] = sum, sums[C+c] = sumsq  (one block / ch)
// ---------------------------------------------------------------------------
__global__ void k_stats(const float* __restrict__ X, float* __restrict__ sums,
                        int M, int C)
{
    __shared__ float s1[256], s2[256];
    const int c = blockIdx.x;
    float a = 0.f, b = 0.f;
    for (int r = threadIdx.x; r < M; r += 256) {
        float v = X[(size_t)r * C + c];
        a += v; b += v * v;
    }
    s1[threadIdx.x] = a; s2[threadIdx.x] = b;
    __syncthreads();
    for (int st = 128; st > 0; st >>= 1) {
        if (threadIdx.x < st) {
            s1[threadIdx.x] += s1[threadIdx.x + st];
            s2[threadIdx.x] += s2[threadIdx.x + st];
        }
        __syncthreads();
    }
    if (threadIdx.x == 0) { sums[c] = s1[0]; sums[C + c] = s2[0]; }
}

// ---------------------------------------------------------------------------
// Fold BN into weights, pack W' to bf16 WMMA B-fragments, compute b'.
// gammaBN == nullptr -> identity (no BN).  Single block of 256 threads.
// ---------------------------------------------------------------------------
__global__ void k_fold(const float* __restrict__ W, const float* __restrict__ bias,
                       const float* __restrict__ gammaBN, const float* __restrict__ betaBN,
                       const float* __restrict__ sums, float invM,
                       float* __restrict__ bp, __bf16* __restrict__ Wfrag, int K)
{
    __shared__ float s[128], t[128];
    const int tid = threadIdx.x;
    if (tid < K) {
        if (gammaBN) {
            float m  = sums[tid] * invM;
            float v  = sums[K + tid] * invM - m * m;
            float sc = gammaBN[tid] * rsqrtf(v + EPS_BN);
            s[tid] = sc; t[tid] = betaBN[tid] - m * sc;
        } else { s[tid] = 1.f; t[tid] = 0.f; }
    }
    __syncthreads();
    if (tid < 64) {
        float acc = bias[tid];
        for (int k = 0; k < K; ++k) acc += t[k] * W[k * 64 + tid];
        bp[tid] = acc;
    }
    const int kcs = K >> 5;
    for (int idx = tid; idx < 64 * K; idx += 256) {
        int e  = idx & 15;
        int l  = (idx >> 4) & 31;
        int ck = (idx >> 9);               // colTile*kcs + c
        int c  = ck % kcs;
        int ct = ck / kcs;
        int col = ct * 16 + (l & 15);
        int kv  = c * 32 + ((e < 8) ? e : e + 8) + ((l >= 16) ? 8 : 0);
        Wfrag[idx] = f2bf(s[kv] * W[kv * 64 + col]);
    }
}

// ---------------------------------------------------------------------------
// elu(H) -> X2[:,0:64];  X2[:,64:128] = 0
// ---------------------------------------------------------------------------
__global__ void k_elu_concat(const float* __restrict__ H, float* __restrict__ X2)
{
    size_t tid = (size_t)blockIdx.x * blockDim.x + threadIdx.x;
    if (tid >= MROWS * 64) return;
    size_t r = tid >> 6; int f = (int)(tid & 63);
    X2[r * 128 + f]      = elu1(H[tid]);
    X2[r * 128 + 64 + f] = 0.f;
}

// ---------------------------------------------------------------------------
// SpMM scatter: X2[:,64:128] += L @ X2[:,0:64]  (per batch, atomic f32 add)
// ---------------------------------------------------------------------------
__global__ void k_spmm(const float* __restrict__ vals, const int* __restrict__ rows,
                       const int* __restrict__ cols, float* __restrict__ X2)
{
    size_t tid = (size_t)blockIdx.x * blockDim.x + threadIdx.x;
    const size_t total = (size_t)BB * NNZ * 64;
    if (tid >= total) return;
    int f = (int)(tid & 63);
    size_t t2 = tid >> 6;
    int i = (int)(t2 % NNZ);
    int b = (int)(t2 / NNZ);
    float v = vals[i] * X2[((size_t)b * NN + cols[i]) * 128 + f];
    float* dst = &X2[((size_t)b * NN + rows[i]) * 128 + 64 + f];
    __hip_atomic_fetch_add(dst, v, __ATOMIC_RELAXED, __HIP_MEMORY_SCOPE_AGENT);
}

// elementwise elu
__global__ void k_elu(const float* __restrict__ X, float* __restrict__ Y, size_t n)
{
    size_t tid = (size_t)blockIdx.x * blockDim.x + threadIdx.x;
    if (tid < n) Y[tid] = elu1(X[tid]);
}

// d[(b*N+n)*64+f] *= l[b*64+f]
__global__ void k_rowscale(float* __restrict__ D, const float* __restrict__ L)
{
    size_t tid = (size_t)blockIdx.x * blockDim.x + threadIdx.x;
    if (tid >= MROWS * 64) return;
    int f = (int)(tid & 63);
    int b = (int)(tid / ((size_t)NN * 64));
    D[tid] *= L[b * 64 + f];
}

// mean over nodes: out[b,f] = mean_n H[b,n,f]  (one block per batch)
__global__ void k_pool(const float* __restrict__ H, float* __restrict__ out)
{
    __shared__ float sm[256];
    const int b = blockIdx.x;
    const int f = threadIdx.x & 63;
    const int g = threadIdx.x >> 6;      // 0..3
    float s = 0.f;
    for (int n = g; n < NN; n += 4) s += H[((size_t)b * NN + n) * 64 + f];
    sm[threadIdx.x] = s;
    __syncthreads();
    if (g == 0) out[b * 64 + f] = (sm[f] + sm[64 + f] + sm[128 + f] + sm[192 + f]) / (float)NN;
}

// generic small dense: Y[row,n] = elu?( b[n] + sum_k X[srcRow,k]*W[k,n] )
__global__ void k_gconv_small(const float* __restrict__ X, const float* __restrict__ W,
                              const float* __restrict__ bias, float* __restrict__ Y,
                              size_t M, int K, int Nc, int bcastMod, int doElu)
{
    size_t tid = (size_t)blockIdx.x * blockDim.x + threadIdx.x;
    if (tid >= M * (size_t)Nc) return;
    size_t row = tid / Nc; int n = (int)(tid % Nc);
    size_t xr = bcastMod > 0 ? (row % (size_t)bcastMod) : row;
    float acc = bias[n];
    for (int k = 0; k < K; ++k) acc += X[xr * K + k] * W[k * Nc + n];
    if (doElu) acc = elu1(acc);
    Y[row * Nc + n] = acc;
}

// z = eps*exp(0.5*logvar)+mu ; copy z/mu/logvar to output
__global__ void k_reparam(const float* __restrict__ eps, const float* __restrict__ mu,
                          const float* __restrict__ lv, float* __restrict__ z_ws,
                          float* __restrict__ oz, float* __restrict__ omu,
                          float* __restrict__ olv)
{
    int i = blockIdx.x * blockDim.x + threadIdx.x;
    if (i >= BB * DLAT) return;
    float z = eps[i] * __expf(0.5f * lv[i]) + mu[i];
    z_ws[i] = z; oz[i] = z; omu[i] = mu[i]; olv[i] = lv[i];
}

__global__ void k_fill_scalar(float* __restrict__ dst, const float* __restrict__ s, size_t n)
{
    size_t tid = (size_t)blockIdx.x * blockDim.x + threadIdx.x;
    if (tid < n) dst[tid] = s[0];
}

// ---------------------------------------------------------------------------
// Host orchestration
// ---------------------------------------------------------------------------
static inline unsigned gridFor(size_t total, int block) {
    return (unsigned)((total + block - 1) / block);
}

struct Ws {
    float* X2;        // MROWS x 128
    float* bufH;      // MROWS x 64
    float* bufT;      // MROWS x 64
    float* sums;      // 256
    float* bp;        // 64
    __bf16* Wfrag;    // 64*128 bf16
    float* smalls;    // pooled/hid/mu/lv/z/lvec/lvec2
};

static void run_resnet(const float* vals, const int* rows, const int* cols,
                       float* cur, float* tmp, const Ws& w,
                       const float* bn0g, const float* bn0b,
                       const float* fc0W, const float* fc0b,
                       const float* bn1g, const float* bn1b,
                       const float* fc1W, const float* fc1b,
                       hipStream_t st)
{
    const float invM = 1.f / (float)MROWS;
    const unsigned gEl  = gridFor(MROWS * 64, 256);
    const unsigned gSp  = gridFor((size_t)BB * NNZ * 64, 256);

    // half 1: h=elu(x); X2=[h, L h]; h1 = BN-fold GEMM(X2)
    k_elu_concat<<<gEl, 256, 0, st>>>(cur, w.X2);
    k_spmm<<<gSp, 256, 0, st>>>(vals, rows, cols, w.X2);
    k_stats<<<128, 256, 0, st>>>(w.X2, w.sums, (int)MROWS, 128);
    k_fold<<<1, 256, 0, st>>>(fc0W, fc0b, bn0g, bn0b, w.sums, invM, w.bp, w.Wfrag, 128);
    launch_gemm(w.X2, w.Wfrag, w.bp, nullptr, tmp, MROWS, 128, false, st);
    // half 2: + residual cur
    k_elu_concat<<<gEl, 256, 0, st>>>(tmp, w.X2);
    k_spmm<<<gSp, 256, 0, st>>>(vals, rows, cols, w.X2);
    k_stats<<<128, 256, 0, st>>>(w.X2, w.sums, (int)MROWS, 128);
    k_fold<<<1, 256, 0, st>>>(fc1W, fc1b, bn1g, bn1b, w.sums, invM, w.bp, w.Wfrag, 128);
    launch_gemm(w.X2, w.Wfrag, w.bp, cur, cur, MROWS, 128, false, st);
}

extern "C" void kernel_launch(void* const* d_in, const int* in_sizes, int n_in,
                              void* d_out, int out_size, void* d_ws, size_t ws_size,
                              hipStream_t stream)
{
    (void)in_sizes; (void)n_in; (void)out_size; (void)ws_size;

    // ---- input indices (setup_inputs dict order, params flattened) ----
    const float* x          = (const float*)d_in[0];
    const float* L_vals     = (const float*)d_in[1];
    const float* meanL_vals = (const float*)d_in[2];
    const float* mean_shape = (const float*)d_in[3];
    const float* eps        = (const float*)d_in[4];
    int p = 5;
    const float* enc_conv1_W = (const float*)d_in[p++];
    const float* enc_conv1_b = (const float*)d_in[p++];
    const float* enc_rn[2][8];
    for (int i = 0; i < 2; ++i)
        for (int j = 0; j < 8; ++j) enc_rn[i][j] = (const float*)d_in[p++];
    const float* enc_bn2_g = (const float*)d_in[p++];
    const float* enc_bn2_b = (const float*)d_in[p++];
    const float* enc_conv2_W = (const float*)d_in[p++];
    const float* enc_conv2_b = (const float*)d_in[p++];
    const float* enc_hidden_W = (const float*)d_in[p++];
    const float* enc_hidden_b = (const float*)d_in[p++];
    const float* enc_mu_W = (const float*)d_in[p++];
    const float* enc_mu_b = (const float*)d_in[p++];
    const float* enc_lv_W = (const float*)d_in[p++];
    const float* enc_lv_b = (const float*)d_in[p++];
    const float* dec_cs_W = (const float*)d_in[p++];
    const float* dec_cs_b = (const float*)d_in[p++];
    const float* dec_cl_W = (const float*)d_in[p++];
    const float* dec_cl_b = (const float*)d_in[p++];
    const float* dec_rn[3][8];
    for (int i = 0; i < 3; ++i)
        for (int j = 0; j < 8; ++j) dec_rn[i][j] = (const float*)d_in[p++];
    const float* dec_d1_W = (const float*)d_in[p++];
    const float* dec_d1_b = (const float*)d_in[p++];
    const float* dec_d2_W = (const float*)d_in[p++];
    const float* dec_d2_b = (const float*)d_in[p++];
    const float* dec_bn2_g = (const float*)d_in[p++];
    const float* dec_bn2_b = (const float*)d_in[p++];
    const float* dec_conv2_W = (const float*)d_in[p++];
    const float* dec_conv2_b = (const float*)d_in[p++];
    const float* dec_mu_W = (const float*)d_in[p++];
    const float* dec_mu_b = (const float*)d_in[p++];
    const float* dec_logvar = (const float*)d_in[p++];
    const int* L_rows = (const int*)d_in[p++];
    const int* L_cols = (const int*)d_in[p++];

    // ---- workspace carve ----
    char* ws = (char*)d_ws;
    Ws w;
    w.X2    = (float*)ws;                 ws += MROWS * 128 * sizeof(float);
    w.bufH  = (float*)ws;                 ws += MROWS * 64 * sizeof(float);
    w.bufT  = (float*)ws;                 ws += MROWS * 64 * sizeof(float);
    w.sums  = (float*)ws;                 ws += 256 * sizeof(float);
    w.bp    = (float*)ws;                 ws += 64 * sizeof(float);
    w.Wfrag = (__bf16*)ws;                ws += 64 * 128 * sizeof(__bf16) + 32;
    w.smalls = (float*)ws;
    float* hpool = w.smalls;              // 64*64
    float* hhid  = hpool + 64 * 64;       // 64*64
    float* muB   = hhid + 64 * 64;        // 512
    float* lvB   = muB + 512;             // 512
    float* zB    = lvB + 512;             // 512
    float* lvec  = zB + 512;              // 64*64
    float* lvec2 = lvec + 64 * 64;        // 64*64

    float* out       = (float*)d_out;
    float* out_rmu   = out;                       // B*N*3
    float* out_rlv   = out + (size_t)BB * NN * 3; // B*N*3
    float* out_z     = out_rlv + (size_t)BB * NN * 3;
    float* out_mu    = out_z + BB * DLAT;
    float* out_lv    = out_mu + BB * DLAT;

    const float invM = 1.f / (float)MROWS;
    const unsigned gEw = gridFor(MROWS * 64, 256);

    // ================= encoder =================
    k_gconv_small<<<gEw, 256, 0, stream>>>(x, enc_conv1_W, enc_conv1_b, w.bufH,
                                           MROWS, 3, 64, 0, 0);
    for (int i = 0; i < 2; ++i)
        run_resnet(L_vals, L_rows, L_cols, w.bufH, w.bufT, w,
                   enc_rn[i][0], enc_rn[i][1], enc_rn[i][2], enc_rn[i][3],
                   enc_rn[i][4], enc_rn[i][5], enc_rn[i][6], enc_rn[i][7], stream);
    // h = elu(h); h = elu(BN-gconv(h, enc_conv2))
    k_elu<<<gEw, 256, 0, stream>>>(w.bufH, w.bufT, MROWS * 64);
    k_stats<<<64, 256, 0, stream>>>(w.bufT, w.sums, (int)MROWS, 64);
    k_fold<<<1, 256, 0, stream>>>(enc_conv2_W, enc_conv2_b, enc_bn2_g, enc_bn2_b,
                                  w.sums, invM, w.bp, w.Wfrag, 64);
    launch_gemm(w.bufT, w.Wfrag, w.bp, nullptr, w.bufH, MROWS, 64, true, stream);
    // pool -> hidden -> mu/logvar -> z
    k_pool<<<BB, 256, 0, stream>>>(w.bufH, hpool);
    k_fold<<<1, 256, 0, stream>>>(enc_hidden_W, enc_hidden_b, nullptr, nullptr,
                                  w.sums, invM, w.bp, w.Wfrag, 64);
    launch_gemm(hpool, w.Wfrag, w.bp, nullptr, hhid, 64, 64, false, stream);
    k_gconv_small<<<2, 256, 0, stream>>>(hhid, enc_mu_W, enc_mu_b, muB, 64, 64, DLAT, 0, 0);
    k_gconv_small<<<2, 256, 0, stream>>>(hhid, enc_lv_W, enc_lv_b, lvB, 64, 64, DLAT, 0, 0);
    k_reparam<<<2, 256, 0, stream>>>(eps, muB, lvB, zB, out_z, out_mu, out_lv);

    // ================= decoder =================
    // d = elu(gconv(broadcast(mean_shape), dec_conv_shape))
    k_gconv_small<<<gEw, 256, 0, stream>>>(mean_shape, dec_cs_W, dec_cs_b, w.bufH,
                                           MROWS, 3, 64, NN, 1);
    // l = dense2(dense1(z))
    k_gconv_small<<<16, 256, 0, stream>>>(zB, dec_d1_W, dec_d1_b, lvec, 64, DLAT, 64, 0, 0);
    k_fold<<<1, 256, 0, stream>>>(dec_d2_W, dec_d2_b, nullptr, nullptr,
                                  w.sums, invM, w.bp, w.Wfrag, 64);
    launch_gemm(lvec, w.Wfrag, w.bp, nullptr, lvec2, 64, 64, false, stream);
    // d = d * l[:,None,:]
    k_rowscale<<<gEw, 256, 0, stream>>>(w.bufH, lvec2);
    // d = elu(gconv(d, dec_conv_latent))   (no BN)
    k_fold<<<1, 256, 0, stream>>>(dec_cl_W, dec_cl_b, nullptr, nullptr,
                                  w.sums, invM, w.bp, w.Wfrag, 64);
    launch_gemm(w.bufH, w.Wfrag, w.bp, nullptr, w.bufT, MROWS, 64, true, stream);
    // 3 decoder resnets (current activation now in bufT)
    for (int i = 0; i < 3; ++i)
        run_resnet(meanL_vals, L_rows, L_cols, w.bufT, w.bufH, w,
                   dec_rn[i][0], dec_rn[i][1], dec_rn[i][2], dec_rn[i][3],
                   dec_rn[i][4], dec_rn[i][5], dec_rn[i][6], dec_rn[i][7], stream);
    // d = elu(d); d = elu(BN-gconv(d, dec_conv2))
    k_elu<<<gEw, 256, 0, stream>>>(w.bufT, w.bufH, MROWS * 64);
    k_stats<<<64, 256, 0, stream>>>(w.bufH, w.sums, (int)MROWS, 64);
    k_fold<<<1, 256, 0, stream>>>(dec_conv2_W, dec_conv2_b, dec_bn2_g, dec_bn2_b,
                                  w.sums, invM, w.bp, w.Wfrag, 64);
    launch_gemm(w.bufH, w.Wfrag, w.bp, nullptr, w.bufT, MROWS, 64, true, stream);
    // recog_mu = gconv(d, dec_mu)  (64->3)
    k_gconv_small<<<gridFor(MROWS * 3, 256), 256, 0, stream>>>(
        w.bufT, dec_mu_W, dec_mu_b, out_rmu, MROWS, 64, 3, 0, 0);
    // recog_logvar = broadcast scalar
    k_fill_scalar<<<gridFor(MROWS * 3, 256), 256, 0, stream>>>(
        out_rlv, dec_logvar, MROWS * 3);
}
